// Attention_2241972929086
// MI455X (gfx1250) — compile-verified, matrix operand order checked by reading, not run on hardware
//
#include <hip/hip_runtime.h>

typedef _Float16 half_t;
typedef __attribute__((ext_vector_type(16))) _Float16 v16h;
typedef __attribute__((ext_vector_type(8)))  _Float16 v8h;
typedef __attribute__((ext_vector_type(8)))  float    v8f;

#define NB 4
#define SS 2048
#define DM 1024
#define NH 16
#define DH 64

__device__ __forceinline__ v8f vzero8() {
  v8f z = {0.f,0.f,0.f,0.f,0.f,0.f,0.f,0.f};
  return z;
}

__device__ __forceinline__ v8f wmma32(v16h a, v16h b, v8f c) {
  // D = A(16x32 f16) * B(32x16 f16) + C(16x16 f32)
  return __builtin_amdgcn_wmma_f32_16x16x32_f16(false, a, false, b, (short)0, c,
                                                false, false);
}

// Per-lane fragment load for the documented 16-bit 16x32 A layout (and its
// mirror for B^T): lane holds row (lane&15); halves a[0..7] = K[8h..8h+7],
// a[8..15] = K[16+8h..23+8h]  (h = lane>>4). Two contiguous 16B loads.
__device__ __forceinline__ v16h load_frag(const half_t* rowbase, int hh) {
  v8h lo = *(const v8h*)(rowbase + 8 * hh);
  v8h hi = *(const v8h*)(rowbase + 16 + 8 * hh);
  return __builtin_shufflevector(lo, hi, 0,1,2,3,4,5,6,7,8,9,10,11,12,13,14,15);
}

// Async global->LDS stage of one 64x32-half W tile (4 KB), cooperatively by
// 128 threads (2 x b128 chunks each). Tracked by ASYNCcnt.
__device__ __forceinline__ void stage_w(half_t* dst_lds, const half_t* WT,
                                        int n0, int k0, int tid) {
  #pragma unroll
  for (int t = 0; t < 2; ++t) {
    int c = tid + 128 * t;        // chunk 0..255, 16B each
    int rw = c >> 2;              // tile row 0..63
    int co = (c & 3) * 8;         // half offset within row (0,8,16,24)
    unsigned lds = (unsigned)(size_t)(dst_lds + rw * 32 + co);
    const half_t* g = WT + (n0 + rw) * DM + k0 + co;
    asm volatile("global_load_async_to_lds_b128 %0, %1, off"
                 :: "v"(lds), "v"(g) : "memory");
  }
}

// ---------------- f32 -> f16 conversions ----------------
__global__ void k_cvt_x(const float* __restrict__ x, half_t* __restrict__ y, int n) {
  int i = blockIdx.x * 256 + threadIdx.x;
  if (i < n) y[i] = (half_t)x[i];
}

// W[k][n] f32 -> WT[n][k] f16  (1024x1024)
__global__ void k_cvt_wt(const float* __restrict__ w, half_t* __restrict__ wt) {
  int i = blockIdx.x * 256 + threadIdx.x;
  int k = i >> 10, n = i & 1023;
  wt[n * DM + k] = (half_t)w[i];
}

// ---------------- QKV projection GEMM ----------------
// C[8192,1024] = Xh[8192,1024] @ W ; per wave: 32(M) x 64(N) tile; W tile
// staged block-wide through LDS via async loads (double-buffered).
// Q,K stored [B,H,S,64]; V stored transposed [B,H,64,S].
__global__ void __launch_bounds__(128)
k_qkv(const half_t* __restrict__ Xh, const half_t* __restrict__ WqT,
      const half_t* __restrict__ WkT, const half_t* __restrict__ WvT,
      half_t* __restrict__ Q, half_t* __restrict__ K, half_t* __restrict__ V) {
  __shared__ __align__(16) half_t wbuf[2][64 * 32];
  const int tid = threadIdx.x;
  const int lane = tid & 31, wid = tid >> 5;
  const int row = lane & 15, hh = lane >> 4;
  const int m0 = (blockIdx.x * 4 + wid) * 32;
  const int h = blockIdx.y, n0 = h * DH;
  const int which = blockIdx.z;
  const half_t* WT = (which == 0) ? WqT : (which == 1) ? WkT : WvT;

  v8f acc[2][4];
  #pragma unroll
  for (int i = 0; i < 2; ++i)
    #pragma unroll
    for (int j = 0; j < 4; ++j) acc[i][j] = vzero8();

  stage_w(&wbuf[0][0], WT, n0, 0, tid);

  for (int k0 = 0; k0 < DM; k0 += 32) {
    const int cur = (k0 >> 5) & 1;
    if (k0 + 32 < DM) {
      stage_w(&wbuf[1 - cur][0], WT, n0, k0 + 32, tid);
      asm volatile("s_wait_asynccnt 0x2" ::: "memory");  // cur batch landed
    } else {
      asm volatile("s_wait_asynccnt 0x0" ::: "memory");
    }
    __syncthreads();

    v16h a0 = load_frag(Xh + (m0 + row) * DM + k0, hh);
    v16h a1 = load_frag(Xh + (m0 + 16 + row) * DM + k0, hh);
    #pragma unroll
    for (int ni = 0; ni < 4; ++ni) {
      v8h lo = *(const v8h*)&wbuf[cur][(ni * 16 + row) * 32 + 8 * hh];
      v8h hi = *(const v8h*)&wbuf[cur][(ni * 16 + row) * 32 + 16 + 8 * hh];
      v16h bf = __builtin_shufflevector(lo, hi,
                    0,1,2,3,4,5,6,7,8,9,10,11,12,13,14,15);
      acc[0][ni] = wmma32(a0, bf, acc[0][ni]);
      acc[1][ni] = wmma32(a1, bf, acc[1][ni]);
    }
    __syncthreads();  // before cur buffer is re-staged next iteration
  }

  const int b = m0 >> 11, s0 = m0 & (SS - 1);
  const int bh = b * NH + h;
  if (which < 2) {
    half_t* dst = (which == 0 ? Q : K) + (bh * SS + s0) * DH;
    #pragma unroll
    for (int mi = 0; mi < 2; ++mi)
      #pragma unroll
      for (int ni = 0; ni < 4; ++ni)
        #pragma unroll
        for (int r = 0; r < 8; ++r)
          dst[(mi * 16 + r + 8 * hh) * DH + ni * 16 + row] = (half_t)acc[mi][ni][r];
  } else {
    #pragma unroll
    for (int mi = 0; mi < 2; ++mi)
      #pragma unroll
      for (int ni = 0; ni < 4; ++ni) {
        v8h pk;
        #pragma unroll
        for (int r = 0; r < 8; ++r) pk[r] = (half_t)acc[mi][ni][r];
        *(v8h*)(V + (bh * DH + ni * 16 + row) * SS + s0 + mi * 16 + 8 * hh) = pk;
      }
  }
}

// ---------------- flash attention, one 16-row q tile per wave ----------------
__global__ void __launch_bounds__(128)
k_attn(const half_t* __restrict__ Q, const half_t* __restrict__ K,
       const half_t* __restrict__ V, const unsigned char* __restrict__ mask_aft,
       half_t* __restrict__ O) {
  const int lane = threadIdx.x & 31, wid = threadIdx.x >> 5;
  const int row = lane & 15, hh = lane >> 4;
  const int qt = blockIdx.x * 4 + wid;
  const int h = blockIdx.y, b = blockIdx.z;
  const int bh = b * NH + h;
  const int qg = qt * 16 + row;  // this lane's q column in the S^T tile

  const half_t* Qb = Q + (bh * SS + qt * 16) * DH;
  v16h qf0 = load_frag(Qb + row * DH, hh);        // B fragment of Q^T, d 0..31
  v16h qf1 = load_frag(Qb + row * DH + 32, hh);   // d 32..63

  v8f o[4];
  #pragma unroll
  for (int dt = 0; dt < 4; ++dt) o[dt] = vzero8();
  float m = -__builtin_inff(), l = 0.f;

  const int nsteps = (qt * 16 + 15) / 32 + 1;  // causal: skip tiles past diag
  for (int st = 0; st < nsteps; ++st) {
    const int kb = st * 32;
    float sc[2][8];
    #pragma unroll
    for (int t = 0; t < 2; ++t) {
      v8f c = vzero8();
      const half_t* Kb = K + (bh * SS + kb + t * 16 + row) * DH;
      c = wmma32(load_frag(Kb, hh), qf0, c);       // S^T = K * Q^T
      c = wmma32(load_frag(Kb + 32, hh), qf1, c);
      #pragma unroll
      for (int r = 0; r < 8; ++r) {
        int kg = kb + t * 16 + r + 8 * hh;
        float s = c[r] * 0.125f;                   // SCALE = 1/sqrt(64)
        sc[t][r] = (kg <= qg) ? s : -__builtin_inff();
      }
    }
    // online softmax over k for fixed q (=lane&15); partner half-wave holds
    // the other 8 k's of each 16-tile.
    float mstep = -__builtin_inff();
    #pragma unroll
    for (int t = 0; t < 2; ++t)
      #pragma unroll
      for (int r = 0; r < 8; ++r) mstep = fmaxf(mstep, sc[t][r]);
    mstep = fmaxf(mstep, __shfl_xor(mstep, 16, 32));
    float mnew = fmaxf(m, mstep);
    float p[2][8], rsum = 0.f;
    #pragma unroll
    for (int t = 0; t < 2; ++t)
      #pragma unroll
      for (int r = 0; r < 8; ++r) {
        p[t][r] = __expf(sc[t][r] - mnew);
        rsum += p[t][r];
      }
    rsum += __shfl_xor(rsum, 16, 32);
    float corr = __expf(m - mnew);
    l = l * corr + rsum;   // denominator uses FULL softmax sum (pre aft-mask)
    m = mnew;
    // post-softmax mask: zero the V contribution only
    #pragma unroll
    for (int t = 0; t < 2; ++t) {
      unsigned long long mb = *(const unsigned long long*)(
          mask_aft + (size_t)qg * SS + kb + t * 16 + 8 * hh);
      #pragma unroll
      for (int r = 0; r < 8; ++r)
        if ((mb >> (8 * r)) & 0xffull) p[t][r] = 0.f;
    }
    // S^T C-layout IS the A-fragment layout for P (16q x 32k): no shuffles
    v16h pa;
    #pragma unroll
    for (int t = 0; t < 2; ++t)
      #pragma unroll
      for (int r = 0; r < 8; ++r) pa[t * 8 + r] = (half_t)p[t][r];
    // rescale running output; O rows are q = r + 8*hh -> fetch that q's corr
    #pragma unroll
    for (int r = 0; r < 8; ++r) {
      float f = __shfl(corr, 8 * hh + r, 32);
      #pragma unroll
      for (int dt = 0; dt < 4; ++dt) o[dt][r] *= f;
    }
    #pragma unroll
    for (int dt = 0; dt < 4; ++dt) {
      v16h vf = load_frag(V + (bh * DH + dt * 16 + row) * SS + kb, hh);
      o[dt] = wmma32(pa, vf, o[dt]);
    }
  }
  // finalize: divide row q by l[q], store [B,S,H*64] f16 for the projection
  #pragma unroll
  for (int r = 0; r < 8; ++r) {
    float linv = 1.f / __shfl(l, 8 * hh + r, 32);
    int q = qt * 16 + r + 8 * hh;
    #pragma unroll
    for (int dt = 0; dt < 4; ++dt)
      O[(size_t)(b * SS + q) * DM + h * DH + dt * 16 + row] =
          (half_t)(o[dt][r] * linv);
  }
}

// ---------------- output projection, f32 result ----------------
__global__ void __launch_bounds__(128)
k_oproj(const half_t* __restrict__ A, const half_t* __restrict__ WT,
        float* __restrict__ Y) {
  __shared__ __align__(16) half_t wbuf[2][64 * 32];
  const int tid = threadIdx.x;
  const int lane = tid & 31, wid = tid >> 5;
  const int row = lane & 15, hh = lane >> 4;
  const int m0 = (blockIdx.x * 4 + wid) * 32;
  const int n0 = blockIdx.y * 64;

  v8f acc[2][4];
  #pragma unroll
  for (int i = 0; i < 2; ++i)
    #pragma unroll
    for (int j = 0; j < 4; ++j) acc[i][j] = vzero8();

  stage_w(&wbuf[0][0], WT, n0, 0, tid);

  for (int k0 = 0; k0 < DM; k0 += 32) {
    const int cur = (k0 >> 5) & 1;
    if (k0 + 32 < DM) {
      stage_w(&wbuf[1 - cur][0], WT, n0, k0 + 32, tid);
      asm volatile("s_wait_asynccnt 0x2" ::: "memory");
    } else {
      asm volatile("s_wait_asynccnt 0x0" ::: "memory");
    }
    __syncthreads();

    v16h a0 = load_frag(A + (size_t)(m0 + row) * DM + k0, hh);
    v16h a1 = load_frag(A + (size_t)(m0 + 16 + row) * DM + k0, hh);
    #pragma unroll
    for (int ni = 0; ni < 4; ++ni) {
      v8h lo = *(const v8h*)&wbuf[cur][(ni * 16 + row) * 32 + 8 * hh];
      v8h hi = *(const v8h*)&wbuf[cur][(ni * 16 + row) * 32 + 16 + 8 * hh];
      v16h bf = __builtin_shufflevector(lo, hi,
                    0,1,2,3,4,5,6,7,8,9,10,11,12,13,14,15);
      acc[0][ni] = wmma32(a0, bf, acc[0][ni]);
      acc[1][ni] = wmma32(a1, bf, acc[1][ni]);
    }
    __syncthreads();
  }
  #pragma unroll
  for (int mi = 0; mi < 2; ++mi)
    #pragma unroll
    for (int ni = 0; ni < 4; ++ni)
      #pragma unroll
      for (int r = 0; r < 8; ++r)
        Y[(size_t)(m0 + mi * 16 + r + 8 * hh) * DM + n0 + ni * 16 + row] =
            acc[mi][ni][r];
}

extern "C" void kernel_launch(void* const* d_in, const int* in_sizes, int n_in,
                              void* d_out, int out_size, void* d_ws, size_t ws_size,
                              hipStream_t stream) {
  const float* Xq = (const float*)d_in[0];
  // d_in[1] = mask_bef (causal) -- recomputed from indices, not read
  const unsigned char* mask_aft = (const unsigned char*)d_in[2];
  const float* Wq = (const float*)d_in[3];
  const float* Wk = (const float*)d_in[4];
  const float* Wv = (const float*)d_in[5];
  const float* Wo = (const float*)d_in[6];
  float* out = (float*)d_out;

  char* ws = (char*)d_ws;
  size_t off = 0;
  const size_t XBYTES = (size_t)NB * SS * DM * sizeof(half_t);   // 16 MB
  const size_t WBYTES = (size_t)DM * DM * sizeof(half_t);        // 2 MB
  half_t* Xh  = (half_t*)(ws + off); off += XBYTES;
  half_t* WqT = (half_t*)(ws + off); off += WBYTES;
  half_t* WkT = (half_t*)(ws + off); off += WBYTES;
  half_t* WvT = (half_t*)(ws + off); off += WBYTES;
  half_t* WoT = (half_t*)(ws + off); off += WBYTES;
  half_t* Qb  = (half_t*)(ws + off); off += XBYTES;
  half_t* Kb  = (half_t*)(ws + off); off += XBYTES;
  half_t* Vt  = (half_t*)(ws + off); off += XBYTES;
  half_t* Att = (half_t*)(ws + off); off += XBYTES;

  const int nX = NB * SS * DM;
  k_cvt_x<<<nX / 256, 256, 0, stream>>>(Xq, Xh, nX);
  k_cvt_wt<<<(DM * DM) / 256, 256, 0, stream>>>(Wq, WqT);
  k_cvt_wt<<<(DM * DM) / 256, 256, 0, stream>>>(Wk, WkT);
  k_cvt_wt<<<(DM * DM) / 256, 256, 0, stream>>>(Wv, WvT);
  k_cvt_wt<<<(DM * DM) / 256, 256, 0, stream>>>(Wo, WoT);

  dim3 gq(NB * SS / 128, NH, 3);
  k_qkv<<<gq, 128, 0, stream>>>(Xh, WqT, WkT, WvT, Qb, Kb, Vt);

  dim3 ga(SS / 16 / 4, NH, NB);
  k_attn<<<ga, 128, 0, stream>>>(Qb, Kb, Vt, mask_aft, Att);

  dim3 go(NB * SS / 128, DM / 64);
  k_oproj<<<go, 128, 0, stream>>>(Att, WoT, out);
}
